// S4Block_37572373905579
// MI455X (gfx1250) — matrix operand order repaired and checked by reading, not compiled, not run on hardware
//
#include <hip/hip_runtime.h>
#include <math.h>

#define TLEN   2048
#define NFFT   4096
#define CDIM   128
#define NSTATE 64
#define BSZ    16

typedef __attribute__((ext_vector_type(16))) __bf16 v16bf;
typedef __attribute__((ext_vector_type(8)))  float  v8f;

static __device__ __forceinline__ float2 cmulf(float2 a, float2 b) {
  return make_float2(a.x * b.x - a.y * b.y, a.x * b.y + a.y * b.x);
}

// round-to-nearest-even float -> bf16 bits
static __device__ __forceinline__ unsigned short f2bf(float f) {
  unsigned u = __float_as_uint(f);
  u += 0x7fffu + ((u >> 16) & 1u);
  return (unsigned short)(u >> 16);
}

// ---------------- FFT helpers (LDS, blockDim.x threads) ----------------
// DIF: natural-order input -> bit-reversed-order output
static __device__ void fft_dif(float2* s, int n, int logn, float sign) {
  for (int p = logn; p >= 1; --p) {
    const int half = 1 << (p - 1);
    const int len  = 1 << p;
    const float ang = sign * 6.283185307179586f / (float)len;
    for (int j = threadIdx.x; j < (n >> 1); j += blockDim.x) {
      const int blk = j >> (p - 1);
      const int pos = j & (half - 1);
      const int i0 = blk * len + pos;
      const int i1 = i0 + half;
      float2 a = s[i0], b = s[i1];
      s[i0] = make_float2(a.x + b.x, a.y + b.y);
      const float dx = a.x - b.x, dy = a.y - b.y;
      float sn, cs;
      __sincosf(ang * (float)pos, &sn, &cs);
      s[i1] = make_float2(dx * cs - dy * sn, dx * sn + dy * cs);
    }
    __syncthreads();
  }
}

// DIT: bit-reversed-order input -> natural-order output
static __device__ void fft_dit(float2* s, int n, int logn, float sign) {
  for (int p = 1; p <= logn; ++p) {
    const int half = 1 << (p - 1);
    const int len  = 1 << p;
    const float ang = sign * 6.283185307179586f / (float)len;
    for (int j = threadIdx.x; j < (n >> 1); j += blockDim.x) {
      const int blk = j >> (p - 1);
      const int pos = j & (half - 1);
      const int i0 = blk * len + pos;
      const int i1 = i0 + half;
      float sn, cs;
      __sincosf(ang * (float)pos, &sn, &cs);
      float2 b = s[i1];
      float2 t = make_float2(b.x * cs - b.y * sn, b.x * sn + b.y * cs);
      float2 a = s[i0];
      s[i1] = make_float2(a.x - t.x, a.y - t.y);
      s[i0] = make_float2(a.x + t.x, a.y + t.y);
    }
    __syncthreads();
  }
}

// ---------------- K0: W_out (o,c) -> Wt bf16 (o,c), row-major by o ----------------
__global__ __launch_bounds__(256) void k_prep_w(const float* __restrict__ W,
                                                unsigned short* __restrict__ Wt) {
  int i = blockIdx.x * 256 + threadIdx.x;   // 128*128
  int c = i >> 7, o = i & 127;
  Wt[o * CDIM + c] = f2bf(W[o * CDIM + c]);
}

// ---------------- K1: LayerNorm, one wave32 per row of 128 ----------------
__global__ __launch_bounds__(256) void k_ln(const float* __restrict__ X,
                                            const float* __restrict__ g,
                                            const float* __restrict__ be,
                                            float* __restrict__ Y) {
  const int wave = threadIdx.x >> 5, lane = threadIdx.x & 31;
  const size_t row = (size_t)blockIdx.x * 8 + wave;
  const float* xr = X + row * CDIM;
  float v0 = xr[lane], v1 = xr[lane + 32], v2 = xr[lane + 64], v3 = xr[lane + 96];
  float sum = v0 + v1 + v2 + v3;
  float sq  = v0 * v0 + v1 * v1 + v2 * v2 + v3 * v3;
  for (int off = 16; off >= 1; off >>= 1) {
    sum += __shfl_xor(sum, off, 32);
    sq  += __shfl_xor(sq,  off, 32);
  }
  const float mu  = sum * (1.0f / 128.0f);
  const float var = sq * (1.0f / 128.0f) - mu * mu;
  const float inv = rsqrtf(var + 1e-5f);
  float* yr = Y + row * CDIM;
  yr[lane]      = (v0 - mu) * inv * g[lane]      + be[lane];
  yr[lane + 32] = (v1 - mu) * inv * g[lane + 32] + be[lane + 32];
  yr[lane + 64] = (v2 - mu) * inv * g[lane + 64] + be[lane + 64];
  yr[lane + 96] = (v3 - mu) * inv * g[lane + 96] + be[lane + 96];
}

// ---------------- K2: at_roots[l,c] via Cauchy sums ----------------
__global__ __launch_bounds__(256) void k_atroots(
    const float* __restrict__ Lr, const float* __restrict__ Li,
    const float* __restrict__ Pr, const float* __restrict__ Pi,
    const float* __restrict__ Br, const float* __restrict__ Bi,
    const float* __restrict__ Cr, const float* __restrict__ Ci,
    float2* __restrict__ AR) {
  const int idx = blockIdx.x * 256 + threadIdx.x;  // TLEN*CDIM
  const int l = idx >> 7, c = idx & 127;
  float sn, cs;
  __sincosf(-6.283185307179586f * (float)l / (float)TLEN, &sn, &cs);
  // omega = cs + i sn
  const float2 om_p = make_float2(1.f + cs,  sn);   // 1 + w
  const float2 om_m = make_float2(1.f - cs, -sn);   // 1 - w
  const float inv_pd = 1.f / (om_p.x * om_p.x + om_p.y * om_p.y);
  float2 g;  // (2/STEP)*(1-w)/(1+w), STEP=0.01
  g.x = 200.f * (om_m.x * om_p.x + om_m.y * om_p.y) * inv_pd;
  g.y = 200.f * (om_m.y * om_p.x - om_m.x * om_p.y) * inv_pd;
  const float2 c2 = make_float2(2.f * om_p.x * inv_pd, -2.f * om_p.y * inv_pd);

  float2 k00 = {0, 0}, k01 = {0, 0}, k10 = {0, 0}, k11 = {0, 0};
  const int base = c * NSTATE;
  for (int n = 0; n < NSTATE; ++n) {
    const float lr = fminf(Lr[base + n], 1e-4f), li = Li[base + n];
    const float pr = Pr[base + n], pi = Pi[base + n];
    const float br = Br[base + n], bi = Bi[base + n];
    const float cr = Cr[base + n], ci = Ci[base + n];
    const float dx = g.x - lr, dy = g.y - li;
    const float invd = 1.f / (dx * dx + dy * dy);
    // numerators
    const float v00x = cr * br + ci * bi, v00y = cr * bi - ci * br;  // conj(C)*B
    const float v01x = cr * pr + ci * pi, v01y = cr * pi - ci * pr;  // conj(C)*P
    const float v10x = pr * br + pi * bi, v10y = pr * bi - pi * br;  // conj(P)*B
    const float v11x = pr * pr + pi * pi;                            // conj(P)*P (real)
    k00.x += (v00x * dx + v00y * dy) * invd;  k00.y += (v00y * dx - v00x * dy) * invd;
    k01.x += (v01x * dx + v01y * dy) * invd;  k01.y += (v01y * dx - v01x * dy) * invd;
    k10.x += (v10x * dx + v10y * dy) * invd;  k10.y += (v10y * dx - v10x * dy) * invd;
    k11.x += (v11x * dx) * invd;              k11.y += (-v11x * dy) * invd;
  }
  const float2 den2 = make_float2(1.f + k11.x, k11.y);
  const float inv2 = 1.f / (den2.x * den2.x + den2.y * den2.y);
  const float2 num = cmulf(k01, k10);
  const float2 frac = make_float2((num.x * den2.x + num.y * den2.y) * inv2,
                                  (num.y * den2.x - num.x * den2.y) * inv2);
  const float2 inner = make_float2(k00.x - frac.x, k00.y - frac.y);
  AR[c * TLEN + l] = cmulf(c2, inner);
}

// ---------------- K3: per-channel kernel build: ifft2048 then fft4096 ----------------
__global__ __launch_bounds__(256) void k_build(const float2* __restrict__ AR,
                                               float2* __restrict__ KF) {
  __shared__ float2 s[NFFT];
  __shared__ float kr[TLEN];
  const int c = blockIdx.x;
  for (int t = threadIdx.x; t < TLEN; t += blockDim.x) s[t] = AR[c * TLEN + t];
  __syncthreads();
  fft_dif(s, TLEN, 11, +1.0f);                      // inverse DFT (unscaled), bitrev out
  const float invL = 1.0f / (float)TLEN;
  for (int t = threadIdx.x; t < TLEN; t += blockDim.x)
    kr[t] = s[__brev((unsigned)t) >> 21].x * invL;  // natural-order real kernel
  __syncthreads();
  for (int t = threadIdx.x; t < NFFT; t += blockDim.x)
    s[t] = make_float2(t < TLEN ? kr[t] : 0.0f, 0.0f);
  __syncthreads();
  fft_dif(s, NFFT, 12, -1.0f);                      // forward, bitrev-order spectrum
  for (int j = threadIdx.x; j < NFFT; j += blockDim.x) KF[c * NFFT + j] = s[j];
}

// ---------------- K4: FFT convolution per (b,c) + D*y + GELU -> bf16 A ----------------
__global__ __launch_bounds__(256) void k_conv(const float* __restrict__ Y,
                                              const float2* __restrict__ KF,
                                              const float* __restrict__ Dv,
                                              unsigned short* __restrict__ H) {
  __shared__ float2 s[NFFT];
  __shared__ float yl[TLEN];
  const int b = blockIdx.x >> 7, c = blockIdx.x & 127;
  const float dc = Dv[c];
  for (int t = threadIdx.x; t < TLEN; t += blockDim.x)
    yl[t] = Y[((size_t)(b * TLEN + t)) * CDIM + c];
  __syncthreads();
  for (int t = threadIdx.x; t < NFFT; t += blockDim.x)
    s[t] = make_float2(t < TLEN ? yl[t] : 0.0f, 0.0f);
  __syncthreads();
  fft_dif(s, NFFT, 12, -1.0f);                      // bitrev spectrum
  for (int j = threadIdx.x; j < NFFT; j += blockDim.x)
    s[j] = cmulf(s[j], KF[c * NFFT + j]);           // bitrev * bitrev, elementwise
  __syncthreads();
  fft_dit(s, NFFT, 12, +1.0f);                      // natural-order time out
  const float invN = 1.0f / (float)NFFT;
  for (int t = threadIdx.x; t < TLEN; t += blockDim.x) {
    const float ssm = s[t].x * invN + dc * yl[t];
    const float h = 0.5f * ssm * (1.0f + erff(ssm * 0.70710678118654752f));
    H[((size_t)(b * TLEN + t)) * CDIM + c] = f2bf(h);
  }
}

// ---------------- K5: WMMA GEMM (32768x128 @ 128x128) + bias + residual ----------------
__global__ __launch_bounds__(256) void k_gemm(const unsigned short* __restrict__ H,
                                              const unsigned short* __restrict__ Wt,
                                              const float* __restrict__ X,
                                              const float* __restrict__ bout,
                                              float* __restrict__ OUT) {
  __shared__ uint4 As4[256];  // 16 rows x 128 bf16 = 4KB
  const unsigned* As_u = (const unsigned*)As4;
  const int tile = blockIdx.x;                       // 2048 tiles of 16 rows
  const uint4* H4 = (const uint4*)H + (size_t)tile * 256;
  As4[threadIdx.x] = H4[threadIdx.x];
  __syncthreads();

  const int wave = threadIdx.x >> 5, lane = threadIdx.x & 31;
  const int m = lane & 15, hi = lane >> 4;
  const int n0 = wave * 16;
  v8f acc = {};
  const uint4* W4 = (const uint4*)Wt;

#pragma unroll
  for (int kc = 0; kc < 4; ++kc) {
    // A fragment: 16-bit A 16x32 layout (VGPR v holds K pair; lane hi half -> +8)
    union { unsigned u[8]; v16bf v; } a;
#pragma unroll
    for (int v_ = 0; v_ < 8; ++v_) {
      const int koff = (v_ < 4 ? 2 * v_ : 16 + 2 * (v_ - 4)) + hi * 8;
      const int k = kc * 32 + koff;                  // even
      a.u[v_] = As_u[m * 64 + (k >> 1)];
    }
    // B fragment: lane holds column N = n0 + (lane&15); K in table order (hi half -> +8)
    const int nrow = n0 + m;
    union { uint4 q[2]; v16bf v; } bf;
    bf.q[0] = W4[nrow * 16 + kc * 4 + hi];
    bf.q[1] = W4[nrow * 16 + kc * 4 + 2 + hi];
    acc = __builtin_amdgcn_wmma_f32_16x16x32_bf16(false, a.v, false, bf.v,
                                                  (short)0, acc, false, false);
  }

  // C/D f32 16x16: VGPR v -> M = v + hi*8, N = lane&15
  const int n = n0 + m;
  const float bb = bout[n];
  const int r0 = tile * 16;
#pragma unroll
  for (int v_ = 0; v_ < 8; ++v_) {
    const size_t idx = (size_t)(r0 + v_ + hi * 8) * CDIM + n;
    OUT[idx] = acc[v_] + bb + X[idx];
  }
}

// ---------------- launch ----------------
extern "C" void kernel_launch(void* const* d_in, const int* in_sizes, int n_in,
                              void* d_out, int out_size, void* d_ws, size_t ws_size,
                              hipStream_t stream) {
  const float* x   = (const float*)d_in[0];
  const float* lng = (const float*)d_in[1];
  const float* lnb = (const float*)d_in[2];
  const float* Lr  = (const float*)d_in[3];
  const float* Li  = (const float*)d_in[4];
  const float* Pr  = (const float*)d_in[5];
  const float* Pi  = (const float*)d_in[6];
  const float* Br  = (const float*)d_in[7];
  const float* Bi  = (const float*)d_in[8];
  const float* Cr  = (const float*)d_in[9];
  const float* Ci  = (const float*)d_in[10];
  const float* Dv  = (const float*)d_in[11];
  const float* W   = (const float*)d_in[12];
  const float* bo  = (const float*)d_in[13];
  float* out = (float*)d_out;

  char* ws = (char*)d_ws;
  size_t off = 0;
  auto take = [&](size_t bytes) -> void* {
    void* p = ws + off;
    off = (off + bytes + 255) & ~(size_t)255;
    return p;
  };
  float*          Y   = (float*)take((size_t)BSZ * TLEN * CDIM * 4);          // 16 MB
  float2*         AR  = (float2*)take((size_t)TLEN * CDIM * 8);               // 2 MB
  float2*         KF  = (float2*)take((size_t)NFFT * CDIM * 8);               // 4 MB
  unsigned short* H   = (unsigned short*)take((size_t)BSZ * TLEN * CDIM * 2); // 8 MB
  unsigned short* Wt  = (unsigned short*)take((size_t)CDIM * CDIM * 2);       // 32 KB
  (void)ws_size; (void)in_sizes; (void)n_in; (void)out_size;

  k_prep_w <<<(CDIM * CDIM) / 256,        256, 0, stream>>>(W, Wt);
  k_ln     <<<(BSZ * TLEN) / 8,           256, 0, stream>>>(x, lng, lnb, Y);
  k_atroots<<<(TLEN * CDIM) / 256,        256, 0, stream>>>(Lr, Li, Pr, Pi, Br, Bi, Cr, Ci, AR);
  k_build  <<<CDIM,                       256, 0, stream>>>(AR, KF);
  k_conv   <<<BSZ * CDIM,                 256, 0, stream>>>(Y, KF, Dv, H);
  k_gemm   <<<(BSZ * TLEN) / 16,          256, 0, stream>>>(H, Wt, x, bo, out);
}
